// FullyInteractedPolynomialPertPredictor_16922171146472
// MI455X (gfx1250) — matrix operand order repaired and checked by reading, not compiled, not run on hardware
//
#include <hip/hip_runtime.h>

typedef __bf16 bf16_t;
typedef __bf16 v16bf __attribute__((ext_vector_type(16)));
typedef __bf16 v8bf  __attribute__((ext_vector_type(8)));
typedef float  v8f   __attribute__((ext_vector_type(8)));

#define B_TOT      1024
#define L_DIM      128
#define P_DIM      32
#define N_DIM      128
#define MBLK       128          // rows of B per workgroup
#define KSTEP      32           // K per WMMA (bf16)
#define KROW_PAD   40           // bf16 elems per n-row in LDS (32 + 8 pad -> 80B stride, bank-conflict free)
#define L_PER_CHUNK 8           // split-K: l-values per chunk
#define STEPS      (L_PER_CHUNK * 32)   // 256 K-steps per chunk
#define N_MBLK     (B_TOT / MBLK)       // 8
#define N_CHUNK    (L_DIM / L_PER_CHUNK) // 16

// fp32 -> bf16 hi/lo split (hi = RN(x), lo = RN(x - hi)); hi*hi+hi*lo+lo*hi ~ fp32
__device__ __forceinline__ void bf16_split(float x, bf16_t& h, bf16_t& l) {
    h = (bf16_t)x;
    l = (bf16_t)(x - (float)h);
}

__global__ void __launch_bounds__(256) init_out_kernel(const float* __restrict__ bias,
                                                       float* __restrict__ out) {
    int idx = blockIdx.x * 256 + threadIdx.x;   // grid 512 * 256 = 131072 = B*N
    out[idx] = bias[idx & (N_DIM - 1)];
}

__global__ void __launch_bounds__(256) fipp_gemm_kernel(
    const float* __restrict__ ctrl,   // (B, L)
    const float* __restrict__ pert,   // (B, P)
    const float* __restrict__ Wg,     // (L*P*P, N)
    float*       __restrict__ out)    // (B, N), pre-initialized with bias
{
    __shared__ __align__(16) bf16_t sWhi[2][N_DIM * KROW_PAD];  // 2 x 10 KB
    __shared__ __align__(16) bf16_t sWlo[2][N_DIM * KROW_PAD];  // 2 x 10 KB
    __shared__ __align__(16) float  sPert[MBLK * 33];           // 16.5 KB (stride 33: conflict-free)

    const int tid    = threadIdx.x;
    const int lane   = tid & 31;
    const int wid    = tid >> 5;                 // 8 waves
    const int mb     = blockIdx.x & (N_MBLK - 1);
    const int chunk  = blockIdx.x >> 3;
    const int bBase  = mb * MBLK;
    const int l0     = chunk * L_PER_CHUNK;

    // ---- stage pert for this M-block into LDS (once) ----
    for (int idx = tid; idx < MBLK * P_DIM; idx += 256) {
        int r = idx >> 5, j = idx & 31;
        sPert[r * 33 + j] = pert[(size_t)(bBase + r) * P_DIM + j];
    }

    // ---- W staging mapping: thread owns column wn, k-half wkh ----
    const int wn  = tid & (N_DIM - 1);
    const int wkh = (tid >> 7) & 1;

    // preload W tile for step 0 (coalesced: consecutive lanes -> consecutive n)
    float wreg[4][4];
    {
        const int k0g = l0 * 1024;   // ((l0*32+0)*32)
        #pragma unroll
        for (int t4 = 0; t4 < 4; ++t4) {
            const int kc = wkh * 4 + t4;
            #pragma unroll
            for (int q = 0; q < 4; ++q)
                wreg[t4][q] = Wg[(size_t)(k0g + kc * 4 + q) * N_DIM + wn];
        }
    }
    __syncthreads();   // sPert ready

    // ---- per-wave A-side persistent state ----
    const int row     = (wid << 4) + (lane & 15);   // 0..127 within M-block
    const int hihalf  = (lane >> 4) & 1;            // lane 16..31 -> 1
    // A 16x32 bf16 VGPR layout: elem e holds K = e + (e>=8 ? 8:0) + 8*hihalf
    float pj[16];
    #pragma unroll
    for (int e = 0; e < 16; ++e) {
        const int kk = e + ((e >= 8) ? 8 : 0) + (hihalf << 3);
        pj[e] = sPert[row * 33 + kk];
    }

    // convert step-0 W tile -> bf16 hi/lo, transposed (n-major) into buffer 0
    #pragma unroll
    for (int t4 = 0; t4 < 4; ++t4) {
        const int kc = wkh * 4 + t4;
        #pragma unroll
        for (int q = 0; q < 4; ++q) {
            bf16_t h, l;
            bf16_split(wreg[t4][q], h, l);
            sWhi[0][wn * KROW_PAD + kc * 4 + q] = h;
            sWlo[0][wn * KROW_PAD + kc * 4 + q] = l;
        }
    }
    __syncthreads();

    v8f acc[8] = {};          // 8 tiles of 16x16 f32 -> 64 VGPRs
    float ctrlv = 0.0f;

    for (int s = 0; s < STEPS; ++s) {
        const int cur = s & 1, nxt = cur ^ 1;
        const int li  = s >> 5, i = s & 31;

        // prefetch next W tile into registers (latency hidden by WMMA phase)
        if (s + 1 < STEPS) {
            const int s1  = s + 1;
            const int k0g = ((l0 + (s1 >> 5)) * 32 + (s1 & 31)) * 32;
            #pragma unroll
            for (int t4 = 0; t4 < 4; ++t4) {
                const int kc = wkh * 4 + t4;
                #pragma unroll
                for (int q = 0; q < 4; ++q)
                    wreg[t4][q] = Wg[(size_t)(k0g + kc * 4 + q) * N_DIM + wn];
            }
        }

        if (i == 0)  // l changes every 32 steps (uniform branch)
            ctrlv = ctrl[(size_t)(bBase + row) * L_DIM + (l0 + li)];

        // ---- build A fragment on the fly: A[m, j] = (ctrl*pert_i) * pert_j ----
        const float sv = ctrlv * sPert[row * 33 + i];
        v16bf ahi, alo;
        #pragma unroll
        for (int e = 0; e < 16; ++e) {
            const float x = sv * pj[e];
            bf16_t h, l;
            bf16_split(x, h, l);
            ahi[e] = h;
            alo[e] = l;
        }

        // ---- 8 N-tiles: B frags from LDS + 3 WMMAs each (bf16x3) ----
        const bf16_t* __restrict__ whiB = &sWhi[cur][0];
        const bf16_t* __restrict__ wloB = &sWlo[cur][0];
        const int kb = hihalf << 3;   // B 32x16: lanes 0-15 hold K0..7,16..23, lanes 16-31 hold K8..15,24..31
        #pragma unroll
        for (int t = 0; t < 8; ++t) {
            const int n = (t << 4) + (lane & 15);
            const v8bf bh0 = *(const v8bf*)(whiB + n * KROW_PAD + kb);
            const v8bf bh1 = *(const v8bf*)(whiB + n * KROW_PAD + kb + 16);
            const v8bf bl0 = *(const v8bf*)(wloB + n * KROW_PAD + kb);
            const v8bf bl1 = *(const v8bf*)(wloB + n * KROW_PAD + kb + 16);
            const v16bf bhi = __builtin_shufflevector(bh0, bh1, 0,1,2,3,4,5,6,7,8,9,10,11,12,13,14,15);
            const v16bf blo = __builtin_shufflevector(bl0, bl1, 0,1,2,3,4,5,6,7,8,9,10,11,12,13,14,15);
            acc[t] = __builtin_amdgcn_wmma_f32_16x16x32_bf16(false, ahi, false, bhi, (short)0, acc[t], false, false);
            acc[t] = __builtin_amdgcn_wmma_f32_16x16x32_bf16(false, ahi, false, blo, (short)0, acc[t], false, false);
            acc[t] = __builtin_amdgcn_wmma_f32_16x16x32_bf16(false, alo, false, bhi, (short)0, acc[t], false, false);
        }

        // convert prefetched tile into the other LDS buffer
        if (s + 1 < STEPS) {
            #pragma unroll
            for (int t4 = 0; t4 < 4; ++t4) {
                const int kc = wkh * 4 + t4;
                #pragma unroll
                for (int q = 0; q < 4; ++q) {
                    bf16_t h, l;
                    bf16_split(wreg[t4][q], h, l);
                    sWhi[nxt][wn * KROW_PAD + kc * 4 + q] = h;
                    sWlo[nxt][wn * KROW_PAD + kc * 4 + q] = l;
                }
            }
        }
        __syncthreads();
    }

    // ---- split-K epilogue: atomic accumulate into bias-initialized output ----
    // C/D layout: VGPR r -> M = r + 8*hihalf ; N = lane&15 within tile
    #pragma unroll
    for (int t = 0; t < 8; ++t) {
        const int col = (t << 4) + (lane & 15);
        #pragma unroll
        for (int r = 0; r < 8; ++r) {
            const int rowg = bBase + (wid << 4) + (hihalf << 3) + r;
            unsafeAtomicAdd(&out[(size_t)rowg * N_DIM + col], acc[t][r]);
        }
    }
}

extern "C" void kernel_launch(void* const* d_in, const int* in_sizes, int n_in,
                              void* d_out, int out_size, void* d_ws, size_t ws_size,
                              hipStream_t stream) {
    const float* ctrl = (const float*)d_in[0];   // (1024, 128)
    const float* pert = (const float*)d_in[1];   // (1024, 32)
    const float* Wg   = (const float*)d_in[2];   // (131072, 128)
    const float* bias = (const float*)d_in[3];   // (128,)
    float* out = (float*)d_out;                  // (1024, 128)

    init_out_kernel<<<dim3(512), dim3(256), 0, stream>>>(bias, out);
    fipp_gemm_kernel<<<dim3(N_MBLK * N_CHUNK), dim3(256), 0, stream>>>(ctrl, pert, Wg, out);
}
